// InvariantPointAttention_43439299232378
// MI455X (gfx1250) — compile-verified, matrix operand order checked by reading, not run on hardware
//
#include <hip/hip_runtime.h>
#include <hip/hip_bf16.h>
#include <math.h>

// Problem dims (fixed by the reference)
#define Hh    12
#define Cc    16
#define PQp   4
#define PVp   8
#define CZc   128
#define CSc   384
#define Bb    2
#define Nn    512
#define Mrows (Bb * Nn)          // 1024
#define FEATW 2112               // H*C + 3*H*PV + H*PV + H*CZ

static constexpr float INV_SQRT3 = 0.57735026918962576f;
static constexpr float W_Cc      = 0.23570226039551584f;   // sqrt(2/(9*PQ))
static constexpr float SCALE_QK  = 0.14433756729740643f;   // (3*C)^-0.5

typedef __attribute__((ext_vector_type(16))) __bf16 v16bf;
typedef __attribute__((ext_vector_type(8)))  float  v8f;

#if defined(__has_builtin)
#if __has_builtin(__builtin_amdgcn_tensor_load_to_lds)
#define HAVE_TDM 1
#endif
#endif
#ifndef HAVE_TDM
#define HAVE_TDM 0
#endif

// CDNA5 WMMA 16x16x32 bf16 lane layout (CDNA5 ISA §7.12.2):
// A 16x32: M = lane&15, K = (e&8 ? 16:0) + 8*(lane>=16) + (e&7)
// B 32x16: N = lane&15, same K striping.
// C/D f32: N = lane&15, reg r -> M = r + 8*(lane>=16).
__device__ __forceinline__ int frag_k(int e, int half) {
  return ((e & 8) ? 16 : 0) + half * 8 + (e & 7);
}

// 8 consecutive floats as two b128 loads
struct F8 { float4 a, b; };
__device__ __forceinline__ F8 ld8(const float* __restrict__ p) {
  F8 r;
  r.a = *(const float4*)(p);
  r.b = *(const float4*)(p + 4);
  return r;
}
__device__ __forceinline__ void pack8(v16bf& f, int off, const F8& s) {
  f[off + 0] = (__bf16)s.a.x; f[off + 1] = (__bf16)s.a.y;
  f[off + 2] = (__bf16)s.a.z; f[off + 3] = (__bf16)s.a.w;
  f[off + 4] = (__bf16)s.b.x; f[off + 5] = (__bf16)s.b.y;
  f[off + 6] = (__bf16)s.b.z; f[off + 7] = (__bf16)s.b.w;
}
__device__ __forceinline__ void pack8s(v16bf& f, int off, const F8& s, float m) {
  f[off + 0] = (__bf16)(s.a.x * m); f[off + 1] = (__bf16)(s.a.y * m);
  f[off + 2] = (__bf16)(s.a.z * m); f[off + 3] = (__bf16)(s.a.w * m);
  f[off + 4] = (__bf16)(s.b.x * m); f[off + 5] = (__bf16)(s.b.y * m);
  f[off + 6] = (__bf16)(s.b.z * m); f[off + 7] = (__bf16)(s.b.w * m);
}

#if HAVE_TDM
// Tensor Data Mover: 2D f32 tile (tile_w elems x tile_h rows, row stride in
// elems) from global -> LDS, optional LDS row padding (CDNA5 ISA D#).
// pad codes: interval 0=2,1=4,2=8,3=16,...,6=128 dwords; amount code+1 dwords.
// 6-arg builtin form (amdgpu-toolchain clang-23):
//   (uint32x4 g0, int32x8 g1, int32x4 g2, int32x4 g3, int32x8, i32 cpol)
typedef __attribute__((ext_vector_type(4))) unsigned int u32x4_t;
typedef __attribute__((ext_vector_type(8))) int i32x8_t;
typedef __attribute__((ext_vector_type(4))) int i32x4_t;

__device__ __forceinline__ void tdm_load_2d(unsigned lds_addr, const void* gptr,
                                            unsigned tile_w, unsigned tile_h,
                                            unsigned stride_elems,
                                            unsigned pad_en, unsigned pad_int_code,
                                            unsigned pad_amt_code) {
  const unsigned long long ga = (unsigned long long)(size_t)gptr;
  u32x4_t g0;
  g0[0] = 1u;                                        // count=1 (valid), user mode
  g0[1] = lds_addr;                                  // LDS byte address
  g0[2] = (unsigned)ga;                              // global_addr[31:0]
  g0[3] = (unsigned)((ga >> 32) & 0x01ffffffu)       // global_addr[56:32]
        | (2u << 30);                                // type=2 ("image")
  i32x8_t g1;
  g1[0] = (int)((2u << 16)                           // data_size=2 -> 4 bytes
        | (pad_en << 20) | (pad_int_code << 22) | (pad_amt_code << 25));
  g1[1] = (int)((tile_w & 0xffffu) << 16);           // tensor_dim0[15:0]
  g1[2] = (int)((tile_w >> 16) | ((tile_h & 0xffffu) << 16));   // dim0 hi | dim1 lo
  g1[3] = (int)((tile_h >> 16) | ((tile_w & 0xffffu) << 16));   // dim1 hi | tile_dim0
  g1[4] = (int)(tile_h & 0xffffu);                   // tile_dim1 (tile_dim2=0)
  g1[5] = (int)stride_elems;                         // tensor_dim0_stride[31:0]
  g1[6] = 0;                                         // stride hi | dim1_stride lo
  g1[7] = 0;                                         // dim1_stride hi
  const i32x4_t gz4 = {0, 0, 0, 0};                  // groups 2/3 unused (2D)
  const i32x8_t gz8 = {0, 0, 0, 0, 0, 0, 0, 0};
  __builtin_amdgcn_tensor_load_to_lds(g0, g1, gz4, gz4, gz8, 0);
}
#endif

// ---------------------------------------------------------------------------
// Generic GEMM with bias:  out[M,Ncols] = A[M,K] @ W[K,Ncols] + bias
// One wave per 16x16 tile. K % 32 == 0, Ncols % 16 == 0 at every call site.
__global__ void k_gemm_bias(const float* __restrict__ A, const float* __restrict__ W,
                            const float* __restrict__ bias, float* __restrict__ out,
                            int K, int Ncols) {
  __shared__ float wt[32][20];   // 20-float row stride: 16B-aligned, no bank conflicts
  const int mt = blockIdx.x * 16;
  const int nt = blockIdx.y * 16;
  const int l = threadIdx.x & 31, half = l >> 4, lo = l & 15;
  const float* arow = A + (size_t)(mt + lo) * K;
  v8f acc = {};
  for (int k0 = 0; k0 < K; k0 += 32) {
    const float* wr = W + (size_t)(k0 + l) * Ncols + nt;
    const float4 w0 = *(const float4*)(wr + 0);
    const float4 w1 = *(const float4*)(wr + 4);
    const float4 w2 = *(const float4*)(wr + 8);
    const float4 w3 = *(const float4*)(wr + 12);
    __syncthreads();                      // single-wave WG: cheap
    *(float4*)&wt[l][0]  = w0;
    *(float4*)&wt[l][4]  = w1;
    *(float4*)&wt[l][8]  = w2;
    *(float4*)&wt[l][12] = w3;
    __syncthreads();
    v16bf af, bf;
    pack8(af, 0, ld8(arow + k0 + half * 8));
    pack8(af, 8, ld8(arow + k0 + 16 + half * 8));
#pragma unroll
    for (int e = 0; e < 16; ++e) bf[e] = (__bf16)wt[frag_k(e, half)][lo];
    acc = __builtin_amdgcn_wmma_f32_16x16x32_bf16(false, af, false, bf,
                                                  (short)0, acc, false, false);
  }
  const float bn = bias[nt + lo];
#pragma unroll
  for (int r = 0; r < 8; ++r)
    out[(size_t)(mt + r + half * 8) * Ncols + nt + lo] = acc[r] + bn;
}

// ---------------------------------------------------------------------------
// Frame rotation+translation of point projections.
__global__ void k_rotate(const float* __restrict__ raw, const float* __restrict__ rot,
                         const float* __restrict__ trans, float* __restrict__ outp, int P) {
  const int idx = blockIdx.x * blockDim.x + threadIdx.x;
  if (idx >= Mrows * P) return;
  const int p = idx % P;
  const int bn = idx / P;
  const float lx = raw[(size_t)bn * 3 * P + 0 * P + p];
  const float ly = raw[(size_t)bn * 3 * P + 1 * P + p];
  const float lz = raw[(size_t)bn * 3 * P + 2 * P + p];
  const float* R = rot + (size_t)bn * 9;
  const float* t = trans + (size_t)bn * 3;
  outp[(size_t)idx * 3 + 0] = R[0] * lx + R[1] * ly + R[2] * lz + t[0];
  outp[(size_t)idx * 3 + 1] = R[3] * lx + R[4] * ly + R[5] * lz + t[1];
  outp[(size_t)idx * 3 + 2] = R[6] * lx + R[7] * ly + R[8] * lz + t[2];
}

// ---------------------------------------------------------------------------
// Pair-bias GEMM (z stream #1): logits[b,h,i,j] = INV_SQRT3*(z@Wb + bb) + mask
// A-tile of z (16 rows x 128) fetched by TDM into LDS (132-dword padded rows:
// bank = (4*lane + k) % 64, conflict-free, 16B-aligned for ds b128 reads).
__global__ void k_bpair(const float* __restrict__ z, const float* __restrict__ Wb,
                        const float* __restrict__ bb, const float* __restrict__ seq_mask,
                        float* __restrict__ logits) {
  const long long row0 = (long long)blockIdx.x * 16;
  const int b  = (int)(row0 / ((long long)Nn * Nn));
  const int i  = (int)((row0 / Nn) % Nn);
  const int j0 = (int)(row0 % Nn);
  const int l = threadIdx.x & 31, half = l >> 4, lo = l & 15;
  const int hcl = (lo < Hh) ? lo : (Hh - 1);      // clamped head, unconditional loads
  const float hm = (lo < Hh) ? 1.0f : 0.0f;
#if HAVE_TDM
  __shared__ float zt[16 * 132];                  // 16 rows, 128+4 pad dwords each
  tdm_load_2d((unsigned)(size_t)&zt[0], z + (size_t)row0 * CZc,
              /*tile_w=*/CZc, /*tile_h=*/16, /*stride=*/CZc,
              /*pad_en=*/1u, /*interval 128dw=*/6u, /*amount 4dw=*/3u);
  __builtin_amdgcn_s_wait_tensorcnt(0);
  asm volatile("" ::: "memory");
  const float* zr = &zt[lo * 132];
#else
  const float* zr = z + (size_t)(row0 + lo) * CZc;
#endif
  v8f acc = {};
  for (int k0 = 0; k0 < CZc; k0 += 32) {
    v16bf af, bf;
    pack8(af, 0, ld8(zr + k0 + half * 8));
    pack8(af, 8, ld8(zr + k0 + 16 + half * 8));
#pragma unroll
    for (int e = 0; e < 16; ++e)
      bf[e] = (__bf16)(Wb[(size_t)(k0 + frag_k(e, half)) * Hh + hcl] * hm);
    acc = __builtin_amdgcn_wmma_f32_16x16x32_bf16(false, af, false, bf,
                                                  (short)0, acc, false, false);
  }
  const int h = lo;
  if (h < Hh) {
    const float smi = seq_mask[b * Nn + i];
    const float bbh = bb[h];
#pragma unroll
    for (int r = 0; r < 8; ++r) {
      const int j = j0 + r + half * 8;
      const float mb = 100000.0f * (smi * seq_mask[b * Nn + j] - 1.0f);
      logits[((size_t)(b * Hh + h) * Nn + i) * Nn + j] =
          INV_SQRT3 * (acc[r] + bbh) + mb;
    }
  }
}

// ---------------------------------------------------------------------------
// q.k^T accumulated onto logits via the WMMA C operand; point-distance term
// fused in the epilogue. K=16 contraction zero-padded to 32.
__global__ void k_qk(const float* __restrict__ q, const float* __restrict__ kvtmp,
                     const float* __restrict__ qp_rot, const float* __restrict__ kvp_rot,
                     const float* __restrict__ gamma, float* __restrict__ logits) {
  const int bh = blockIdx.x;            // b*H + h
  const int b = bh / Hh, h = bh % Hh;
  const int i0 = blockIdx.y * 16, j0 = blockIdx.z * 16;
  const int l = threadIdx.x & 31, half = l >> 4, lo = l & 15;

  v8f acc;                              // C = existing logits tile
#pragma unroll
  for (int r = 0; r < 8; ++r)
    acc[r] = logits[((size_t)bh * Nn + i0 + r + half * 8) * Nn + j0 + lo];

  v16bf af = {}, bf = {};               // elems 8..15 stay zero (K padding)
  pack8s(af, 0, ld8(q + (size_t)(b * Nn + i0 + lo) * (Hh * Cc) + h * Cc + half * 8),
         SCALE_QK);
  pack8(bf, 0, ld8(kvtmp + (size_t)(b * Nn + j0 + lo) * (Hh * 2 * Cc) + h * 2 * Cc + half * 8));
  acc = __builtin_amdgcn_wmma_f32_16x16x32_bf16(false, af, false, bf,
                                                (short)0, acc, false, false);

  const float g = gamma[h];
  const float headw = (g > 0.f) ? (g + log1pf(expf(-g))) : log1pf(expf(g)); // softplus
  const float pfac = -0.5f * headw * W_Cc * INV_SQRT3;

  const int j = j0 + lo;
  const float* kpp = kvp_rot + (size_t)(b * Nn + j) * 432 + h * 36;
  const float4 kp0 = *(const float4*)(kpp + 0);
  const float4 kp1 = *(const float4*)(kpp + 4);
  const float4 kp2 = *(const float4*)(kpp + 8);

#pragma unroll
  for (int r = 0; r < 8; ++r) {
    const int i = i0 + r + half * 8;
    const float* qpp = qp_rot + (size_t)(b * Nn + i) * 144 + h * 12;
    const float4 q0 = *(const float4*)(qpp + 0);
    const float4 q1 = *(const float4*)(qpp + 4);
    const float4 q2 = *(const float4*)(qpp + 8);
    float ss = 0.f;
    { float d;
      d = q0.x - kp0.x; ss += d * d;  d = q0.y - kp0.y; ss += d * d;
      d = q0.z - kp0.z; ss += d * d;  d = q0.w - kp0.w; ss += d * d;
      d = q1.x - kp1.x; ss += d * d;  d = q1.y - kp1.y; ss += d * d;
      d = q1.z - kp1.z; ss += d * d;  d = q1.w - kp1.w; ss += d * d;
      d = q2.x - kp2.x; ss += d * d;  d = q2.y - kp2.y; ss += d * d;
      d = q2.z - kp2.z; ss += d * d;  d = q2.w - kp2.w; ss += d * d;
    }
    logits[((size_t)bh * Nn + i) * Nn + j] = acc[r] + pfac * ss;
  }
}

// ---------------------------------------------------------------------------
// In-place row softmax: one wave32 per row of 512, 16 elems/lane.
__global__ void k_softmax(float* __restrict__ logits) {
  float* row = logits + (size_t)blockIdx.x * Nn;
  const int l = threadIdx.x;
  float v[16];
  float mx = -3.0e38f;
#pragma unroll
  for (int e = 0; e < 16; ++e) { v[e] = row[e * 32 + l]; mx = fmaxf(mx, v[e]); }
#pragma unroll
  for (int s = 16; s >= 1; s >>= 1) mx = fmaxf(mx, __shfl_xor(mx, s, 32));
  float sum = 0.f;
#pragma unroll
  for (int e = 0; e < 16; ++e) { v[e] = __expf(v[e] - mx); sum += v[e]; }
#pragma unroll
  for (int s = 16; s >= 1; s >>= 1) sum += __shfl_xor(sum, s, 32);
  const float inv = 1.0f / sum;
#pragma unroll
  for (int e = 0; e < 16; ++e) row[e * 32 + l] = v[e] * inv;
}

// ---------------------------------------------------------------------------
// o = a@v and op = a@vp. sel 0: o (16 C cols); sel 1/2: op cols 0..15 / 16..23.
__global__ void k_oav(const float* __restrict__ probs, const float* __restrict__ kvtmp,
                      const float* __restrict__ kvp_rot, float* __restrict__ feat,
                      float* __restrict__ op_buf) {
  const int bh = blockIdx.x;
  const int b = bh / Hh, h = bh % Hh;
  const int i0 = blockIdx.y * 16;
  const int sel = blockIdx.z;
  const int l = threadIdx.x & 31, half = l >> 4, lo = l & 15;
  const float* arow = probs + ((size_t)bh * Nn + i0 + lo) * Nn;

  const int col = (sel - 1) * 16 + lo;                 // sel>0 path, valid < 24
  const int ccl = (col < PVp * 3) ? col : (PVp * 3 - 1);
  const float cm = (col < PVp * 3) ? 1.0f : 0.0f;
  const int p = ccl / 3, x = ccl % 3;

  v8f acc = {};
  for (int k0 = 0; k0 < Nn; k0 += 32) {
    v16bf af, bf;
    pack8(af, 0, ld8(arow + k0 + half * 8));
    pack8(af, 8, ld8(arow + k0 + 16 + half * 8));
#pragma unroll
    for (int e = 0; e < 16; ++e) {
      const int kk = k0 + frag_k(e, half);    // j index
      float bv;
      if (sel == 0)
        bv = kvtmp[((size_t)(b * Nn) + kk) * (Hh * 2 * Cc) + h * 2 * Cc + Cc + lo];
      else
        bv = kvp_rot[((size_t)(b * Nn + kk) * 144 + h * 12 + PQp + p) * 3 + x] * cm;
      bf[e] = (__bf16)bv;
    }
    acc = __builtin_amdgcn_wmma_f32_16x16x32_bf16(false, af, false, bf,
                                                  (short)0, acc, false, false);
  }
#pragma unroll
  for (int r = 0; r < 8; ++r) {
    const int i = i0 + r + half * 8;
    if (sel == 0) {
      feat[(size_t)(b * Nn + i) * FEATW + h * Cc + lo] = acc[r];
    } else if (col < PVp * 3) {
      op_buf[(((size_t)(b * Nn + i) * Hh + h) * PVp + p) * 3 + x] = acc[r];
    }
  }
}

// ---------------------------------------------------------------------------
// Inverse rotation of op, norms, scatter into feat segments.
__global__ void k_finalize(const float* __restrict__ op_buf, const float* __restrict__ rot,
                           const float* __restrict__ trans, float* __restrict__ feat) {
  const int idx = blockIdx.x * blockDim.x + threadIdx.x;  // over M*H*PV
  if (idx >= Mrows * Hh * PVp) return;
  const int hp = idx % (Hh * PVp);
  const int bn = idx / (Hh * PVp);
  const float* R = rot + (size_t)bn * 9;
  const float* t = trans + (size_t)bn * 3;
  const float dx = op_buf[(size_t)idx * 3 + 0] - t[0];
  const float dy = op_buf[(size_t)idx * 3 + 1] - t[1];
  const float dz = op_buf[(size_t)idx * 3 + 2] - t[2];
  const float ox = R[0] * dx + R[3] * dy + R[6] * dz;   // R^T * d
  const float oy = R[1] * dx + R[4] * dy + R[7] * dz;
  const float oz = R[2] * dx + R[5] * dy + R[8] * dz;
  const float nr = fmaxf(sqrtf(ox * ox + oy * oy + oz * oz), 1e-6f);
  float* f = feat + (size_t)bn * FEATW;
  f[192 + hp] = ox;
  f[288 + hp] = oy;
  f[384 + hp] = oz;
  f[480 + hp] = nr;
}

// ---------------------------------------------------------------------------
// opair = einsum('bhij,bijc->bihc') — z stream #2. Per (b,i): M=heads(pad 16),
// K=j(512), N=c tile of 16. z B-tiles double-buffered through LDS by the TDM,
// s_wait_tensorcnt(1) overlaps DMA of the next tile with the current WMMA.
__global__ void k_opair(const float* __restrict__ probs, const float* __restrict__ z,
                        float* __restrict__ feat) {
  const int bi = blockIdx.x;   // b*N + i
  const int b = bi / Nn, i = bi % Nn;
  const int c0 = blockIdx.y * 16;
  const int l = threadIdx.x & 31, half = l >> 4, lo = l & 15;
  const int hrow = (lo < Hh) ? lo : (Hh - 1);
  const float hm = (lo < Hh) ? 1.0f : 0.0f;
  const float* arow = probs + ((size_t)(b * Hh + hrow) * Nn + i) * Nn;
  const float* zrow = z + (size_t)bi * Nn * CZc;
  v8f acc = {};
#if HAVE_TDM
  __shared__ float zt[2][32 * 20];   // 32 rows x (16 + 4 pad) dwords, conflict-free
  auto kstep = [&](int k0, const float* zb) {
    v16bf af, bf;
    pack8s(af, 0, ld8(arow + k0 + half * 8), hm);
    pack8s(af, 8, ld8(arow + k0 + 16 + half * 8), hm);
#pragma unroll
    for (int e = 0; e < 16; ++e)
      bf[e] = (__bf16)zb[frag_k(e, half) * 20 + lo];
    acc = __builtin_amdgcn_wmma_f32_16x16x32_bf16(false, af, false, bf,
                                                  (short)0, acc, false, false);
  };
  tdm_load_2d((unsigned)(size_t)&zt[0][0], zrow + c0,
              /*tile_w=*/16, /*tile_h=*/32, /*stride=*/CZc,
              /*pad_en=*/1u, /*interval 16dw=*/3u, /*amount 4dw=*/3u);
  for (int k0 = 0; k0 < Nn - 32; k0 += 32) {
    const int nxt = ((k0 >> 5) + 1) & 1;
    tdm_load_2d((unsigned)(size_t)&zt[nxt][0], zrow + (size_t)(k0 + 32) * CZc + c0,
                16, 32, CZc, 1u, 3u, 3u);
    __builtin_amdgcn_s_wait_tensorcnt(1);
    asm volatile("" ::: "memory");
    kstep(k0, &zt[(k0 >> 5) & 1][0]);
  }
  __builtin_amdgcn_s_wait_tensorcnt(0);
  asm volatile("" ::: "memory");
  kstep(Nn - 32, &zt[((Nn - 32) >> 5) & 1][0]);
#else
  for (int k0 = 0; k0 < Nn; k0 += 32) {
    if (k0 + 32 < Nn) __builtin_prefetch(zrow + (size_t)(k0 + 32) * CZc + c0, 0, 1);
    v16bf af, bf;
    pack8s(af, 0, ld8(arow + k0 + half * 8), hm);
    pack8s(af, 8, ld8(arow + k0 + 16 + half * 8), hm);
#pragma unroll
    for (int e = 0; e < 16; ++e)
      bf[e] = (__bf16)zrow[(size_t)(k0 + frag_k(e, half)) * CZc + c0 + lo];
    acc = __builtin_amdgcn_wmma_f32_16x16x32_bf16(false, af, false, bf,
                                                  (short)0, acc, false, false);
  }
#endif
#pragma unroll
  for (int r = 0; r < 8; ++r) {
    const int h = r + half * 8;
    if (h < Hh)
      feat[(size_t)bi * FEATW + 576 + h * CZc + c0 + lo] = acc[r];
  }
}

// ---------------------------------------------------------------------------
extern "C" void kernel_launch(void* const* d_in, const int* in_sizes, int n_in,
                              void* d_out, int out_size, void* d_ws, size_t ws_size,
                              hipStream_t stream) {
  const float* s        = (const float*)d_in[0];
  const float* z        = (const float*)d_in[1];
  const float* rot      = (const float*)d_in[2];
  const float* trans    = (const float*)d_in[3];
  const float* seq_mask = (const float*)d_in[4];
  const float* Wq    = (const float*)d_in[5];
  const float* bq    = (const float*)d_in[6];
  const float* Wkv   = (const float*)d_in[7];
  const float* bkv   = (const float*)d_in[8];
  const float* Wb    = (const float*)d_in[9];
  const float* bb    = (const float*)d_in[10];
  const float* Wqp   = (const float*)d_in[11];
  const float* bqp   = (const float*)d_in[12];
  const float* Wkvp  = (const float*)d_in[13];
  const float* bkvp  = (const float*)d_in[14];
  const float* gamma = (const float*)d_in[15];
  const float* Wout  = (const float*)d_in[16];
  const float* bout  = (const float*)d_in[17];
  float* out = (float*)d_out;

  float* ws = (float*)d_ws;
  float* qtmp    = ws;                    // 1024*192
  float* kvtmp   = qtmp    + 196608;      // 1024*384
  float* qptmp   = kvtmp   + 393216;      // 1024*144
  float* kvptmp  = qptmp   + 147456;      // 1024*432
  float* qp_rot  = kvptmp  + 442368;      // 1024*48*3
  float* kvp_rot = qp_rot  + 147456;      // 1024*144*3
  float* logits  = kvp_rot + 442368;      // 2*12*512*512 (probs in place)
  float* op_buf  = logits  + 6291456;     // 1024*12*8*3
  float* feat    = op_buf  + 294912;      // 1024*2112

  const dim3 wv(32);
  // Projections from s (K = 384)
  k_gemm_bias<<<dim3(64, 12), wv, 0, stream>>>(s, Wq,   bq,   qtmp,   CSc, Hh * Cc);
  k_gemm_bias<<<dim3(64, 24), wv, 0, stream>>>(s, Wkv,  bkv,  kvtmp,  CSc, Hh * 2 * Cc);
  k_gemm_bias<<<dim3(64,  9), wv, 0, stream>>>(s, Wqp,  bqp,  qptmp,  CSc, 3 * PQp * Hh);
  k_gemm_bias<<<dim3(64, 27), wv, 0, stream>>>(s, Wkvp, bkvp, kvptmp, CSc, 3 * (PQp + PVp) * Hh);
  // Frame transforms of points
  k_rotate<<<(Mrows * 48  + 255) / 256, 256, 0, stream>>>(qptmp,  rot, trans, qp_rot, 48);
  k_rotate<<<(Mrows * 144 + 255) / 256, 256, 0, stream>>>(kvptmp, rot, trans, kvp_rot, 144);
  // Logits: pair bias + mask (z stream #1), then qk + point term accumulated
  k_bpair<<<Bb * Nn * Nn / 16, wv, 0, stream>>>(z, Wb, bb, seq_mask, logits);
  k_qk<<<dim3(Bb * Hh, Nn / 16, Nn / 16), wv, 0, stream>>>(qtmp, kvtmp, qp_rot, kvp_rot,
                                                           gamma, logits);
  k_softmax<<<Bb * Hh * Nn, wv, 0, stream>>>(logits);
  // Outputs: o / op, then inverse rotation + norms, then opair (z stream #2)
  k_oav<<<dim3(Bb * Hh, Nn / 16, 3), wv, 0, stream>>>(logits, kvtmp, kvp_rot, feat, op_buf);
  k_finalize<<<(Mrows * Hh * PVp + 255) / 256, 256, 0, stream>>>(op_buf, rot, trans, feat);
  k_opair<<<dim3(Bb * Nn, CZc / 16), wv, 0, stream>>>(logits, z, feat);
  // Final projection feat(1024x2112) @ Wout(2112x384) + bout
  k_gemm_bias<<<dim3(64, 24), wv, 0, stream>>>(feat, Wout, bout, out, FEATW, CSc);
}
